// AttentionBlock_54125177864610
// MI455X (gfx1250) — matrix operand order repaired and checked by reading, not compiled, not run on hardware
//
#include <hip/hip_runtime.h>

// ---------------------------------------------------------------------------
// Types for CDNA5 WMMA (wave32): v_wmma_f32_16x16x32_bf16
// ---------------------------------------------------------------------------
typedef __attribute__((ext_vector_type(16))) __bf16 v16bf;
typedef __attribute__((ext_vector_type(8)))  float  v8f;
typedef __attribute__((ext_vector_type(4)))  int    v4i;

union FragAB { v16bf v; unsigned u[8]; };

__device__ __forceinline__ unsigned short f2bf(float f) {
  unsigned u = __float_as_uint(f);
  u += 0x7FFFu + ((u >> 16) & 1u);          // round-to-nearest-even
  return (unsigned short)(u >> 16);
}
__device__ __forceinline__ unsigned packbf2(float a, float b) {
  return (unsigned)f2bf(a) | ((unsigned)f2bf(b) << 16);
}
__device__ __forceinline__ v8f vzero8() {
  v8f z;
#pragma unroll
  for (int i = 0; i < 8; ++i) z[i] = 0.f;
  return z;
}

// ---------------------------------------------------------------------------
// CDNA5 async global->LDS staging (ASYNCcnt); sync fallback if not declared.
// Builtin params are v4i pointers (16B chunk), AS1 src / AS3 dst.
// ---------------------------------------------------------------------------
#if __has_builtin(__builtin_amdgcn_global_load_async_to_lds_b128)
#define HAS_ASYNC_LDS 1
#endif

__device__ __forceinline__ void stage16(const unsigned* __restrict__ g,
                                        unsigned* l) {
#ifdef HAS_ASYNC_LDS
  __builtin_amdgcn_global_load_async_to_lds_b128(
      (__attribute__((address_space(1))) v4i*)g,
      (__attribute__((address_space(3))) v4i*)l, 0, 0);
#else
  *(uint4*)l = *(const uint4*)g;
#endif
}

__device__ __forceinline__ void wait_stage() {
#ifdef HAS_ASYNC_LDS
#if __has_builtin(__builtin_amdgcn_s_wait_asynccnt)
  __builtin_amdgcn_s_wait_asynccnt(0);
#else
  asm volatile("s_wait_asynccnt 0x0" ::: "memory");
#endif
#endif
}

// ---------------------------------------------------------------------------
// Constants: T=64, B=1, C=768, spatial=8*8*8=512, heads=12, hd=64, groups=12
// ---------------------------------------------------------------------------
#define NTOK   32768      // T*B*S
#define CCH    768
#define C3     2304
#define SPAT   512
#define NHEAD  12
#define HD     64
#define ASTR   20         // LDS row stride in u32 (16B aligned, conflict-free)
#define QSTR   68         // attention LDS row stride in floats (16B aligned)

// ---------------------------------------------------------------------------
// Kernel 0: prep — cast w_in / w_out to bf16, build T5 rel-pos bias table
// ---------------------------------------------------------------------------
__global__ __launch_bounds__(256)
void prep_kernel(const float* __restrict__ w_in, const float* __restrict__ w_out,
                 const float* __restrict__ rel_emb,
                 unsigned short* __restrict__ w_in_b,
                 unsigned short* __restrict__ w_out_b,
                 float* __restrict__ biasT) {
  const int n1 = C3 * CCH;        // 1,769,472
  const int n2 = CCH * CCH;       //   589,824
  const int n3 = NHEAD * 64 * 64; //    49,152
  int i = blockIdx.x * 256 + threadIdx.x;
  if (i < n1) {
    w_in_b[i] = f2bf(w_in[i]);
  } else if (i < n1 + n2) {
    int j = i - n1;
    w_out_b[j] = f2bf(w_out[j]);
  } else if (i < n1 + n2 + n3) {
    int j = i - n1 - n2;
    int h  = j >> 12;         // head
    int rr = (j >> 6) & 63;   // context (row)
    int cc = j & 63;          // memory  (col)
    int nrel = rr - cc;       // n = ctx - mem
    int ret = (nrel < 0) ? 16 : 0;   // NUM_BUCKETS/2
    int na = nrel < 0 ? -nrel : nrel;
    int bucket;
    if (na < 8) {                     // max_exact = 8
      bucket = ret + na;
    } else {
      int vl = 8 + (int)(logf((float)na * 0.125f) * (8.0f / logf(16.0f)));
      if (vl > 15) vl = 15;
      bucket = ret + vl;
    }
    biasT[h * 4096 + rr * 64 + cc] = rel_emb[bucket * NHEAD + h];
  }
}

// ---------------------------------------------------------------------------
// Kernel 1: GroupNorm (weight-only) -> bf16 activations, token-major [NTOK,768]
// ---------------------------------------------------------------------------
__global__ __launch_bounds__(256)
void groupnorm_kernel(const float* __restrict__ x, const float* __restrict__ norm1_w,
                      unsigned short* __restrict__ xn) {
  const int t = blockIdx.x, g = blockIdx.y, tid = threadIdx.x;
  const float* src = x + (size_t)t * (CCH * SPAT) + (size_t)g * (64 * SPAT);
  float s = 0.f, ss = 0.f;
#pragma unroll 4
  for (int k = 0; k < 128; ++k) {
    float v = src[tid + k * 256];
    s += v; ss += v * v;
  }
  __shared__ float r1[256], r2[256];
  r1[tid] = s; r2[tid] = ss;
  __syncthreads();
  for (int off = 128; off > 0; off >>= 1) {
    if (tid < off) { r1[tid] += r1[tid + off]; r2[tid] += r2[tid + off]; }
    __syncthreads();
  }
  const float mu  = r1[0] * (1.f / 32768.f);
  const float var = r2[0] * (1.f / 32768.f) - mu * mu;
  const float inv = rsqrtf(var + 1e-5f);
  for (int k = 0; k < 128; ++k) {
    int li = tid + k * 256;
    int cl = li >> 9, sp = li & 511;
    int c = g * 64 + cl;
    float v = (src[li] - mu) * inv * norm1_w[c];
    xn[((size_t)(t * SPAT + sp)) * CCH + c] = f2bf(v);
  }
}

// ---------------------------------------------------------------------------
// Kernel 2/4: bf16 WMMA GEMM  C[m,n] = sum_k A[m,k]*B[n,k] (+ epilogue)
//   A: [32768, 768] bf16 row-major; B: [N, 768] bf16 row-major (u32 pairs)
//   mode 0: C = qkv fp32 [m, 2304], add b_in
//   mode 1: out[t,c,s] = x + gamma[c]*(C + b_out)  (residual scatter)
// Block tile 128x128, 8 waves -> 32x64 (2x4 WMMA tiles) per wave.
// Double-buffered LDS; async global->LDS staging on the CDNA5 path.
// ---------------------------------------------------------------------------
__global__ __launch_bounds__(256)
void gemm_kernel(const unsigned* __restrict__ A32, const unsigned* __restrict__ B32,
                 const float* __restrict__ biasv, int mode,
                 float* __restrict__ C, const float* __restrict__ x,
                 const float* __restrict__ gamma) {
  __shared__ __align__(16) unsigned as32[2][128 * ASTR];
  __shared__ __align__(16) unsigned bs32[2][128 * ASTR];

  const int tid = threadIdx.x;
  const int m0 = blockIdx.x * 128;
  const int n0 = blockIdx.y * 128;
  const int lane = tid & 31, w = tid >> 5;
  const int half16 = (lane >> 4) & 1, lm = lane & 15;
  const int wr = (w & 3) * 32, wc = (w >> 2) * 64;

  // stage one 128x32-bf16 tile each of A and B: 512 16B-chunks per tile
  auto stage_tiles = [&](int kc, int buf) {
#pragma unroll
    for (int i = 0; i < 2; ++i) {
      int lin = i * 256 + tid;            // chunk id: row = lin/4, cu = lin%4
      int row = lin >> 2, cu = lin & 3;
      stage16(A32 + (size_t)(m0 + row) * 384 + kc * 16 + cu * 4,
              &as32[buf][row * ASTR + cu * 4]);
    }
#pragma unroll
    for (int i = 0; i < 2; ++i) {
      int lin = i * 256 + tid;
      int row = lin >> 2, cu = lin & 3;
      stage16(B32 + (size_t)(n0 + row) * 384 + kc * 16 + cu * 4,
              &bs32[buf][row * ASTR + cu * 4]);
    }
  };

  v8f acc[2][4];
#pragma unroll
  for (int i = 0; i < 2; ++i)
#pragma unroll
    for (int j = 0; j < 4; ++j) acc[i][j] = vzero8();

  stage_tiles(0, 0);
  wait_stage();
  __syncthreads();

  for (int kc = 0; kc < 24; ++kc) {        // K = 768 = 24 * 32
    const int cur = kc & 1;
    if (kc + 1 < 24) stage_tiles(kc + 1, cur ^ 1);

    FragAB a[2], b[4];
#pragma unroll
    for (int i = 0; i < 2; ++i) {          // A frag: M on lanes, K pairs on regs
      int m = wr + 16 * i + lm;
#pragma unroll
      for (int r = 0; r < 8; ++r)
        a[i].u[r] =
            as32[cur][m * ASTR + ((r >= 4) ? 8 : 0) + (half16 ? 4 : 0) + (r & 3)];
    }
#pragma unroll
    for (int j = 0; j < 4; ++j) {          // B frag: N on lanes, same K order as A
      int nn = wc + 16 * j + lm;
#pragma unroll
      for (int r = 0; r < 8; ++r)
        b[j].u[r] = bs32[cur][nn * ASTR + ((r >= 4) ? 8 : 0) + (half16 ? 4 : 0) + (r & 3)];
    }
#pragma unroll
    for (int i = 0; i < 2; ++i)
#pragma unroll
      for (int j = 0; j < 4; ++j)
        acc[i][j] = __builtin_amdgcn_wmma_f32_16x16x32_bf16(
            false, a[i].v, false, b[j].v, (short)0, acc[i][j], false, false);

    wait_stage();        // next buffer resident in LDS (this wave's copies)
    __syncthreads();     // everyone done computing cur + staging next
  }

  // epilogue (C/D layout: N=lane%16, M=reg + 8*(lane>=16))
#pragma unroll
  for (int i = 0; i < 2; ++i) {
#pragma unroll
    for (int j = 0; j < 4; ++j) {
#pragma unroll
      for (int r = 0; r < 8; ++r) {
        int row = m0 + wr + 16 * i + r + half16 * 8;
        int col = n0 + wc + 16 * j + lm;
        float val = acc[i][j][r] + biasv[col];
        if (mode == 0) {
          C[(size_t)row * C3 + col] = val;
        } else {
          int t = row >> 9, sp = row & 511;
          size_t oidx = (size_t)t * (CCH * SPAT) + (size_t)col * SPAT + sp;
          C[oidx] = x[oidx] + gamma[col] * val;
        }
      }
    }
  }
}

// ---------------------------------------------------------------------------
// Kernel 3: attention per (voxel n, head he): 64x64x64 with q/k layernorm,
// T5 bias, softmax. 128 threads = 4 waves; wave w owns rows [16w, 16w+16).
// ---------------------------------------------------------------------------
__global__ __launch_bounds__(128)
void attn_kernel(const float* __restrict__ qkv,
                 const float* __restrict__ qn_w, const float* __restrict__ qn_b,
                 const float* __restrict__ kn_w, const float* __restrict__ kn_b,
                 const float* __restrict__ biasT,
                 unsigned short* __restrict__ obuf) {
  __shared__ __align__(16) float q_s[64 * QSTR];
  __shared__ __align__(16) float k_s[64 * QSTR];
  __shared__ __align__(16) float v_s[64 * QSTR];

  const int n = blockIdx.x, he = blockIdx.y;
  const int tid = threadIdx.x;
  const float* base = qkv + (size_t)n * C3 + he * (3 * HD);

  // stage q/k/v (each 64x64 fp32 = 1024 16B-chunks), row t strided by SPAT*C3
#pragma unroll
  for (int part = 0; part < 3; ++part) {
    float* dst = part == 0 ? q_s : part == 1 ? k_s : v_s;
    const float* src = base + part * HD;
#pragma unroll
    for (int i = 0; i < 8; ++i) {
      int lin = i * 128 + tid;
      int row = lin >> 4, c4 = (lin & 15) << 2;
      stage16((const unsigned*)(src + (size_t)row * (SPAT * C3)) + c4,
              (unsigned*)&dst[row * QSTR + c4]);
    }
  }
  wait_stage();
  __syncthreads();

  // layernorm: threads 0..63 -> q rows (fold 1/sqrt(hd)); 64..127 -> k rows
  {
    int r = tid & 63;
    float* mat = (tid < 64) ? q_s : k_s;
    const float* wv = (tid < 64) ? qn_w : kn_w;
    const float* bv = (tid < 64) ? qn_b : kn_b;
    float scl = (tid < 64) ? 0.125f : 1.0f;
    float s = 0.f;
    for (int j = 0; j < 64; ++j) s += mat[r * QSTR + j];
    float mu = s * (1.f / 64.f);
    float ss = 0.f;
    for (int j = 0; j < 64; ++j) { float d = mat[r * QSTR + j] - mu; ss += d * d; }
    float inv = rsqrtf(ss * (1.f / 64.f) + 1e-5f);
    for (int j = 0; j < 64; ++j) {
      float d = mat[r * QSTR + j] - mu;
      mat[r * QSTR + j] = (d * inv * wv[j] + bv[j]) * scl;
    }
  }
  __syncthreads();

  const int lane = tid & 31, w = tid >> 5;
  const int half16 = (lane >> 4) & 1, lm = lane & 15;

  // S = Q * K^T : wave w computes rows [16w,16w+16) x 4 column tiles
  v8f accS[4];
#pragma unroll
  for (int jt = 0; jt < 4; ++jt) accS[jt] = vzero8();
  for (int kk = 0; kk < 2; ++kk) {
    FragAB a;
    int m = 16 * w + lm;
#pragma unroll
    for (int r = 0; r < 8; ++r) {
      int ko = kk * 32 + ((r >= 4) ? 16 : 0) + (half16 ? 8 : 0) + 2 * (r & 3);
      a.u[r] = packbf2(q_s[m * QSTR + ko], q_s[m * QSTR + ko + 1]);
    }
#pragma unroll
    for (int jt = 0; jt < 4; ++jt) {
      FragAB b;                              // B[c,j] = K[j,c]
      int jr = jt * 16 + lm;
#pragma unroll
      for (int r = 0; r < 8; ++r) {
        int kcx = kk * 32 + ((r >= 4) ? 16 : 0) + (half16 ? 8 : 0) + 2 * (r & 3);
        b.u[r] = packbf2(k_s[jr * QSTR + kcx], k_s[jr * QSTR + kcx + 1]);
      }
      accS[jt] = __builtin_amdgcn_wmma_f32_16x16x32_bf16(
          false, a.v, false, b.v, (short)0, accS[jt], false, false);
    }
  }

  // bias + softmax (row = 16w + r + 8*half16, shared across the 16-lane half)
  const float* bh = biasT + (size_t)he * 4096;
  float p[4][8];
#pragma unroll
  for (int r = 0; r < 8; ++r) {
    int row = 16 * w + r + half16 * 8;
    float mx = -3.0e38f;
#pragma unroll
    for (int jt = 0; jt < 4; ++jt) {
      int col = jt * 16 + lm;
      p[jt][r] = accS[jt][r] + bh[row * 64 + col];
      mx = fmaxf(mx, p[jt][r]);
    }
    for (int msk = 1; msk < 16; msk <<= 1) mx = fmaxf(mx, __shfl_xor(mx, msk, 32));
    float sum = 0.f;
#pragma unroll
    for (int jt = 0; jt < 4; ++jt) { p[jt][r] = __expf(p[jt][r] - mx); sum += p[jt][r]; }
    for (int msk = 1; msk < 16; msk <<= 1) sum += __shfl_xor(sum, msk, 32);
    float invs = 1.f / sum;
#pragma unroll
    for (int jt = 0; jt < 4; ++jt) {
      int col = jt * 16 + lm;
      q_s[row * QSTR + col] = p[jt][r] * invs;  // reuse q_s for P (wave-local rows)
    }
  }
  // (no barrier: P rows are wave-local; LDS is in-order within a wave)

  // O = P * V
  v8f accO[4];
#pragma unroll
  for (int ct = 0; ct < 4; ++ct) accO[ct] = vzero8();
  for (int kk = 0; kk < 2; ++kk) {
    FragAB a;
    int m = 16 * w + lm;
#pragma unroll
    for (int r = 0; r < 8; ++r) {
      int ko = kk * 32 + ((r >= 4) ? 16 : 0) + (half16 ? 8 : 0) + 2 * (r & 3);
      a.u[r] = packbf2(q_s[m * QSTR + ko], q_s[m * QSTR + ko + 1]);
    }
#pragma unroll
    for (int ct = 0; ct < 4; ++ct) {
      FragAB b;                              // B[j,c] = V[j,c]
      int cv = ct * 16 + lm;
#pragma unroll
      for (int r = 0; r < 8; ++r) {
        int j0 = kk * 32 + ((r >= 4) ? 16 : 0) + (half16 ? 8 : 0) + 2 * (r & 3);
        b.u[r] = packbf2(v_s[j0 * QSTR + cv], v_s[(j0 + 1) * QSTR + cv]);
      }
      accO[ct] = __builtin_amdgcn_wmma_f32_16x16x32_bf16(
          false, a.v, false, b.v, (short)0, accO[ct], false, false);
    }
  }

  // write O as bf16 in [token, 768] layout for the output GEMM
#pragma unroll
  for (int ct = 0; ct < 4; ++ct) {
#pragma unroll
    for (int r = 0; r < 8; ++r) {
      int trow = 16 * w + r + half16 * 8;
      int col = ct * 16 + lm;
      obuf[((size_t)trow * SPAT + n) * CCH + he * HD + col] = f2bf(accO[ct][r]);
    }
  }
}

// ---------------------------------------------------------------------------
// Launcher
// ---------------------------------------------------------------------------
extern "C" void kernel_launch(void* const* d_in, const int* in_sizes, int n_in,
                              void* d_out, int out_size, void* d_ws, size_t ws_size,
                              hipStream_t stream) {
  const float* x       = (const float*)d_in[0];
  const float* norm1_w = (const float*)d_in[1];
  const float* w_in    = (const float*)d_in[2];
  const float* b_in    = (const float*)d_in[3];
  const float* qn_w    = (const float*)d_in[4];
  const float* qn_b    = (const float*)d_in[5];
  const float* kn_w    = (const float*)d_in[6];
  const float* kn_b    = (const float*)d_in[7];
  const float* rel_emb = (const float*)d_in[8];
  const float* w_out   = (const float*)d_in[9];
  const float* b_out   = (const float*)d_in[10];
  const float* gamma   = (const float*)d_in[11];
  float* out = (float*)d_out;

  // workspace layout (256B-aligned offsets)
  char* ws = (char*)d_ws;
  unsigned short* xnb   = (unsigned short*)(ws + 0);           //  50,331,648 B
  unsigned short* winb  = (unsigned short*)(ws + 50331648);    //   3,538,944 B
  unsigned short* woutb = (unsigned short*)(ws + 53870592);    //   1,179,648 B
  float*          biasT = (float*)        (ws + 55050240);     //     196,608 B
  float*          qkv   = (float*)        (ws + 55246848);     // 301,989,888 B
  unsigned short* obuf  = (unsigned short*)(ws + 357236736);   //  50,331,648 B

  prep_kernel<<<dim3(9408), dim3(256), 0, stream>>>(w_in, w_out, rel_emb,
                                                    winb, woutb, biasT);
  groupnorm_kernel<<<dim3(64, 12), dim3(256), 0, stream>>>(x, norm1_w, xnb);
  gemm_kernel<<<dim3(NTOK / 128, C3 / 128), dim3(256), 0, stream>>>(
      (const unsigned*)xnb, (const unsigned*)winb, b_in, 0, qkv, nullptr, nullptr);
  attn_kernel<<<dim3(SPAT, NHEAD), dim3(128), 0, stream>>>(
      qkv, qn_w, qn_b, kn_w, kn_b, biasT, obuf);
  gemm_kernel<<<dim3(NTOK / 128, CCH / 128), dim3(256), 0, stream>>>(
      (const unsigned*)obuf, (const unsigned*)woutb, b_out, 1, out, x, gamma);
}
